// SemanticGraphNetwork_1297080123539
// MI455X (gfx1250) — matrix operand order repaired and checked by reading, not compiled, not run on hardware
//
#include <hip/hip_runtime.h>

// ---------------------------------------------------------------------------
// Types & helpers
// ---------------------------------------------------------------------------
typedef __attribute__((ext_vector_type(16))) __bf16 v16bf;
typedef __attribute__((ext_vector_type(8)))  float  v8f;

__device__ __forceinline__ __bf16 f2bf(float x) {
  union { float f; unsigned u; } a; a.f = x;
  unsigned r = a.u + 0x7FFFu + ((a.u >> 16) & 1u);   // round-to-nearest-even
  unsigned short h = (unsigned short)(r >> 16);
  return __builtin_bit_cast(__bf16, h);
}

__device__ __forceinline__ v8f wmma_bf16(v16bf a, v16bf b, v8f c) {
  return __builtin_amdgcn_wmma_f32_16x16x32_bf16(false, a, false, b, (short)0, c,
                                                 false, false);
}

// Async global->LDS copy, 16B per lane, tracked by ASYNCcnt (CDNA5).
// LDS aperture flat addresses carry the LDS byte offset in addr[31:0].
__device__ __forceinline__ void async_ld_b128(void* lds_ptr, const void* gptr) {
  unsigned loff = (unsigned)(size_t)lds_ptr;
  asm volatile("global_load_async_to_lds_b128 %0, %1, off"
               :: "v"(loff), "v"(gptr) : "memory");
}
__device__ __forceinline__ void wait_async0() {
  asm volatile("s_wait_asynccnt 0x0" ::: "memory");
}

// A-matrix fragment (16x32 bf16): lane m(0..15)=row; VGPR halves split K at 8:
//   lanes 0-15 hold K=[0..7],[16..23]; lanes 16-31 hold K=[8..15],[24..31]
__device__ __forceinline__ v16bf load_afrag(const __bf16* base, int stride,
                                            int r0, int k0) {
  int lane = threadIdx.x & 31;
  const __bf16* p = base + (size_t)(r0 + (lane & 15)) * stride + k0 + ((lane >> 4) * 8);
  union { v16bf v; uint4 q[2]; } u;
  u.q[0] = *(const uint4*)p;
  u.q[1] = *(const uint4*)(p + 16);
  return u.v;
}

// B-matrix fragment (32x16 bf16), stored N-major (Bt[n][k]): lane n(0..15)=col;
//   lanes 0-15 hold K=[0..7],[16..23]; lanes 16-31 hold K=[8..15],[24..31]
__device__ __forceinline__ v16bf load_bfrag(const __bf16* base, int stride,
                                            int n0, int k0) {
  int lane = threadIdx.x & 31;
  const __bf16* p = base + (size_t)(n0 + (lane & 15)) * stride + k0 + ((lane >> 4) * 8);
  union { v16bf v; uint4 q[2]; } u;
  u.q[0] = *(const uint4*)p;
  u.q[1] = *(const uint4*)(p + 16);
  return u.v;
}

// ---------------------------------------------------------------------------
// Pre-conversion kernels (run once per launch; removes per-tile f32->bf16 work)
// ---------------------------------------------------------------------------
__global__ void f32_to_bf16_kernel(const float* __restrict__ in,
                                   __bf16* __restrict__ out, size_t n) {
  size_t i = ((size_t)blockIdx.x * blockDim.x + threadIdx.x) * 4;
  if (i + 3 < n) {
    float4 v = *(const float4*)(in + i);
    out[i + 0] = f2bf(v.x); out[i + 1] = f2bf(v.y);
    out[i + 2] = f2bf(v.z); out[i + 3] = f2bf(v.w);
  }
}

// Wt[n][k] = bf16(W[k][n])  (pre-transpose so B tiles are contiguous copies)
__global__ void transpose_bf16_kernel(const float* __restrict__ W,
                                      __bf16* __restrict__ Wt, int K, int N) {
  size_t j = (size_t)blockIdx.x * blockDim.x + threadIdx.x;
  if (j < (size_t)K * N) {
    int n = (int)(j / K), k = (int)(j % K);
    Wt[j] = f2bf(W[(size_t)k * N + n]);
  }
}

// ---------------------------------------------------------------------------
// bf16 WMMA GEMM v2:  C = act(A @ W + bias),  A/Wt pre-converted bf16,
// Wt pre-transposed [N][K].  Tile 128x128, BK=32, 256 threads = 8 waves;
// wave -> 32 rows x 64 cols (8 WMMA / K-step).  Double-buffered LDS staged
// with global_load_async_to_lds_b128 overlapped with WMMA compute.
// GATHER=1 builds A rows as concat(nodeH[src], nodeH[dst]) (bf16 gather).
// ---------------------------------------------------------------------------
template <int GATHER>
__global__ __launch_bounds__(256, 2)
void gemm_bf16_v2(const __bf16* __restrict__ A, const __bf16* __restrict__ Wt,
                  const float* __restrict__ bias, float* __restrict__ C,
                  int M, int K, int Ncols, int do_relu,
                  const long long* __restrict__ srcIdx,
                  const long long* __restrict__ dstIdx,
                  const __bf16* __restrict__ nodeH) {
  __shared__ __align__(16) __bf16 As[2][128][40];   // [buf][row][k]
  __shared__ __align__(16) __bf16 Ws[2][128][40];   // [buf][n][k]

  const int tid = threadIdx.x;
  const int lane = tid & 31;
  const int w = tid >> 5;
  const int m0 = blockIdx.y * 128;
  const int n0 = blockIdx.x * 128;
  const int wr = (w & 3) * 32;      // wave row offset   (2 row-blocks)
  const int wc = (w >> 2) * 64;     // wave col offset   (4 col-blocks)

  // Each wave issues 2 async b128 per matrix per tile:
  // unit u = (w*2+i)*32 + lane ; row = u>>2 ; c8 = (u&3)*8  (4 chunks/row)
  auto issue_tile = [&](int k0, int buf) {
    #pragma unroll
    for (int i = 0; i < 2; ++i) {
      int u = (w * 2 + i) * 32 + lane;
      int r = u >> 2, c8 = (u & 3) * 8;
      const __bf16* ga;
      if (GATHER) {
        long long nid = (k0 < 512) ? srcIdx[m0 + r] : dstIdx[m0 + r];
        ga = nodeH + (size_t)nid * 512 + ((k0 + c8) & 511);
      } else {
        ga = A + (size_t)(m0 + r) * K + k0 + c8;
      }
      async_ld_b128(&As[buf][r][c8], ga);
      async_ld_b128(&Ws[buf][r][c8], Wt + (size_t)(n0 + r) * K + k0 + c8);
    }
  };

  v8f acc[2][4] = {};

  issue_tile(0, 0);
  const int nk = K >> 5;
  for (int kt = 0; kt < nk; ++kt) {
    wait_async0();          // this wave's copies for buf (kt&1) landed
    __syncthreads();        // everyone's copies landed; prev compute finished
    if (kt + 1 < nk) issue_tile((kt + 1) * 32, (kt + 1) & 1);  // prefetch
    const int b = kt & 1;
    v16bf af0 = load_afrag(&As[b][0][0], 40, wr, 0);
    v16bf af1 = load_afrag(&As[b][0][0], 40, wr + 16, 0);
    #pragma unroll
    for (int cb = 0; cb < 4; ++cb) {
      v16bf bf = load_bfrag(&Ws[b][0][0], 40, wc + cb * 16, 0);
      acc[0][cb] = wmma_bf16(af0, bf, acc[0][cb]);
      acc[1][cb] = wmma_bf16(af1, bf, acc[1][cb]);
    }
  }

  const int nloc = lane & 15;
  const int mhi = (lane >> 4) * 8;
  #pragma unroll
  for (int rb2 = 0; rb2 < 2; ++rb2)
    #pragma unroll
    for (int cb = 0; cb < 4; ++cb)
      #pragma unroll
      for (int e = 0; e < 8; ++e) {
        int Mr = m0 + wr + rb2 * 16 + mhi + e;
        int Nc = n0 + wc + cb * 16 + nloc;
        float v = acc[rb2][cb][e] + bias[Nc];
        if (do_relu) v = fmaxf(v, 0.f);
        C[(size_t)Mr * Ncols + Nc] = v;
      }
}

// ---------------------------------------------------------------------------
// In-place LayerNorm over rows of width 512; one wave per row.
// ---------------------------------------------------------------------------
__global__ __launch_bounds__(256)
void ln_kernel(float* __restrict__ X, const float* __restrict__ g,
               const float* __restrict__ b, int rows) {
  int row = blockIdx.x * 8 + (threadIdx.x >> 5);
  int lane = threadIdx.x & 31;
  if (row >= rows) return;
  float* x = X + (size_t)row * 512;
  float v[16], s = 0.f, ss = 0.f;
  #pragma unroll
  for (int j = 0; j < 16; ++j) {
    v[j] = x[lane + 32 * j];
    s += v[j]; ss += v[j] * v[j];
  }
  #pragma unroll
  for (int off = 1; off < 32; off <<= 1) {
    s  += __shfl_xor(s,  off, 32);
    ss += __shfl_xor(ss, off, 32);
  }
  float mu = s * (1.f / 512.f);
  float var = ss * (1.f / 512.f) - mu * mu;
  float r = rsqrtf(var + 1e-5f);
  #pragma unroll
  for (int j = 0; j < 16; ++j) {
    int c = lane + 32 * j;
    x[c] = (v[j] - mu) * r * g[c] + b[c];
  }
}

// ---------------------------------------------------------------------------
// Relation classifier second layer: out[e,0:6] = rc_hidden[e,:] @ w2 + b2
// ---------------------------------------------------------------------------
__global__ __launch_bounds__(256)
void relcls_kernel(const float* __restrict__ h, const float* __restrict__ w2,
                   const float* __restrict__ b2, float* __restrict__ out) {
  int e = blockIdx.x * 8 + (threadIdx.x >> 5);
  int lane = threadIdx.x & 31;
  const float* hr = h + (size_t)e * 256;
  float acc[6] = {0.f, 0.f, 0.f, 0.f, 0.f, 0.f};
  for (int k = lane; k < 256; k += 32) {
    float hv = hr[k];
    #pragma unroll
    for (int j = 0; j < 6; ++j) acc[j] += hv * w2[k * 6 + j];
  }
  #pragma unroll
  for (int j = 0; j < 6; ++j)
    #pragma unroll
    for (int off = 1; off < 32; off <<= 1) acc[j] += __shfl_xor(acc[j], off, 32);
  if (lane < 6) out[(size_t)e * 6 + lane] = acc[lane] + b2[lane];
}

__global__ void zero_kernel(float* __restrict__ p, size_t n) {
  size_t i = (size_t)blockIdx.x * blockDim.x + threadIdx.x;
  if (i < n) p[i] = 0.f;
}

// ---------------------------------------------------------------------------
// Masked two-pass flash attention (unchanged from v1 — exact softmax + exact
// head-mean output).  grid = (N/64 q-tiles, 2 head-groups); 4 waves/block,
// one wave per head.  Scale 0.125 folded into bf16 Q conversion (exact).
// ---------------------------------------------------------------------------
#define QSTR 264   // 256 + 8 pad (bf16)
#define VSTR 72    // 64 + 8 pad (bf16)

__global__ __launch_bounds__(128, 1)
void attn_kernel(const float* __restrict__ qkv, const int* __restrict__ adj,
                 float* __restrict__ attended, float* __restrict__ attn_mean) {
  __shared__ __align__(16) __bf16 Ks[64 * QSTR];          // K tile  [key][c]
  __shared__ __align__(16) __bf16 QV[256 * VSTR];         // Qs then Vt (union)
  __shared__ __align__(16) __bf16 Ptile[4][64][VSTR];     // per-wave P tile
  __shared__ float accS[64 * 64];                         // 4-head prob sum
  __shared__ float mrow[4][64], lrow[4][64];

  const int tid = threadIdx.x;
  const int lane = tid & 31;
  const int w = tid >> 5;
  const int hg = blockIdx.y;
  const int q0 = blockIdx.x * 64;
  const int colbase = hg * 256;
  const int nloc = lane & 15;
  const int mhi = (lane >> 4) * 8;

  __bf16* Qs = QV;
  for (int t = tid; t < 4096; t += 128) {
    int r = t >> 6, c4 = (t & 63) * 4;
    float4 v = *(const float4*)(qkv + (size_t)(q0 + r) * 1536 + colbase + c4);
    Qs[r * QSTR + c4 + 0] = f2bf(v.x * 0.125f);
    Qs[r * QSTR + c4 + 1] = f2bf(v.y * 0.125f);
    Qs[r * QSTR + c4 + 2] = f2bf(v.z * 0.125f);
    Qs[r * QSTR + c4 + 3] = f2bf(v.w * 0.125f);
  }
  __syncthreads();

  v16bf qf[4][2];
  #pragma unroll
  for (int rbq = 0; rbq < 4; ++rbq)
    #pragma unroll
    for (int kc = 0; kc < 2; ++kc)
      qf[rbq][kc] = load_afrag(Qs, QSTR, rbq * 16, w * 64 + kc * 32);
  __syncthreads();

  for (int i = tid; i < 4 * 64; i += 128) {
    (&mrow[0][0])[i] = -1e30f;
    (&lrow[0][0])[i] = 0.f;
  }
  __syncthreads();

  // ======================= PASS 1: softmax stats =======================
  for (int kt = 0; kt < 64; ++kt) {
    const int k0 = kt * 64;
    __syncthreads();
    for (int t = tid; t < 4096; t += 128) {
      int r = t >> 6, c4 = (t & 63) * 4;
      float4 v = *(const float4*)(qkv + (size_t)(k0 + r) * 1536 + 512 + colbase + c4);
      Ks[r * QSTR + c4 + 0] = f2bf(v.x); Ks[r * QSTR + c4 + 1] = f2bf(v.y);
      Ks[r * QSTR + c4 + 2] = f2bf(v.z); Ks[r * QSTR + c4 + 3] = f2bf(v.w);
    }
    __syncthreads();

    v8f s[4][4] = {};
    #pragma unroll
    for (int cb = 0; cb < 4; ++cb) {
      #pragma unroll
      for (int kc = 0; kc < 2; ++kc) {
        v16bf bf = load_bfrag(Ks, QSTR, cb * 16, w * 64 + kc * 32);
        #pragma unroll
        for (int rbq = 0; rbq < 4; ++rbq)
          s[rbq][cb] = wmma_bf16(qf[rbq][kc], bf, s[rbq][cb]);
      }
    }
    #pragma unroll
    for (int rbq = 0; rbq < 4; ++rbq) {
      #pragma unroll
      for (int e = 0; e < 8; ++e) {
        int Mq = q0 + rbq * 16 + mhi + e;
        float vmax = -3e38f;
        #pragma unroll
        for (int cb = 0; cb < 4; ++cb) {
          int Nk = k0 + cb * 16 + nloc;
          float sv = s[rbq][cb][e];
          sv = adj[(size_t)Mq * 4096 + Nk] ? sv : -1e9f;
          s[rbq][cb][e] = sv;
          vmax = fmaxf(vmax, sv);
        }
        #pragma unroll
        for (int off = 1; off < 16; off <<= 1)
          vmax = fmaxf(vmax, __shfl_xor(vmax, off, 32));
        int ridx = rbq * 16 + mhi + e;
        float mold = mrow[w][ridx];
        float mnew = fmaxf(mold, vmax);
        float pe = 0.f;
        #pragma unroll
        for (int cb = 0; cb < 4; ++cb) pe += __expf(s[rbq][cb][e] - mnew);
        #pragma unroll
        for (int off = 1; off < 16; off <<= 1) pe += __shfl_xor(pe, off, 32);
        if (nloc == 0) {
          lrow[w][ridx] = lrow[w][ridx] * __expf(mold - mnew) + pe;
          mrow[w][ridx] = mnew;
        }
      }
    }
  }
  __syncthreads();

  // ======================= PASS 2: probabilities + P@V =================
  __bf16* Vt = QV;
  v8f av[4][4] = {};

  for (int kt = 0; kt < 64; ++kt) {
    const int k0 = kt * 64;
    __syncthreads();
    for (int t = tid; t < 4096; t += 128) {
      int r = t >> 6, c4 = (t & 63) * 4;
      float4 v = *(const float4*)(qkv + (size_t)(k0 + r) * 1536 + 512 + colbase + c4);
      Ks[r * QSTR + c4 + 0] = f2bf(v.x); Ks[r * QSTR + c4 + 1] = f2bf(v.y);
      Ks[r * QSTR + c4 + 2] = f2bf(v.z); Ks[r * QSTR + c4 + 3] = f2bf(v.w);
    }
    for (int t = tid; t < 4096; t += 128) {
      int r = t >> 6, c4 = (t & 63) * 4;
      float4 v = *(const float4*)(qkv + (size_t)(k0 + r) * 1536 + 1024 + colbase + c4);
      Vt[(c4 + 0) * VSTR + r] = f2bf(v.x);
      Vt[(c4 + 1) * VSTR + r] = f2bf(v.y);
      Vt[(c4 + 2) * VSTR + r] = f2bf(v.z);
      Vt[(c4 + 3) * VSTR + r] = f2bf(v.w);
    }
    for (int i = tid; i < 4096; i += 128) accS[i] = 0.f;
    __syncthreads();

    v8f s[4][4] = {};
    #pragma unroll
    for (int cb = 0; cb < 4; ++cb) {
      #pragma unroll
      for (int kc = 0; kc < 2; ++kc) {
        v16bf bf = load_bfrag(Ks, QSTR, cb * 16, w * 64 + kc * 32);
        #pragma unroll
        for (int rbq = 0; rbq < 4; ++rbq)
          s[rbq][cb] = wmma_bf16(qf[rbq][kc], bf, s[rbq][cb]);
      }
    }
    #pragma unroll
    for (int rbq = 0; rbq < 4; ++rbq) {
      #pragma unroll
      for (int e = 0; e < 8; ++e) {
        int Mq = q0 + rbq * 16 + mhi + e;
        int ridx = rbq * 16 + mhi + e;
        float mv = mrow[w][ridx];
        float lv = lrow[w][ridx];
        float inv = (lv > 0.f) ? (1.f / lv) : 0.f;
        #pragma unroll
        for (int cb = 0; cb < 4; ++cb) {
          int cidx = cb * 16 + nloc;
          float sv = s[rbq][cb][e];
          sv = adj[(size_t)Mq * 4096 + k0 + cidx] ? sv : -1e9f;
          float p = __expf(sv - mv) * inv;
          atomicAdd(&accS[ridx * 64 + cidx], p);
          Ptile[w][ridx][cidx] = f2bf(p);
        }
      }
    }
    __syncthreads();
    for (int i = tid; i < 4096; i += 128) {
      int r = i >> 6, c = i & 63;
      atomicAdd(&attn_mean[(size_t)(q0 + r) * 4096 + k0 + c], accS[i] * 0.125f);
    }
    #pragma unroll
    for (int kc = 0; kc < 2; ++kc) {
      v16bf pf[4];
      #pragma unroll
      for (int rbq = 0; rbq < 4; ++rbq)
        pf[rbq] = load_afrag(&Ptile[w][0][0], VSTR, rbq * 16, kc * 32);
      #pragma unroll
      for (int dcb = 0; dcb < 4; ++dcb) {
        v16bf vf = load_bfrag(Vt, VSTR, w * 64 + dcb * 16, kc * 32);
        #pragma unroll
        for (int rbq = 0; rbq < 4; ++rbq)
          av[rbq][dcb] = wmma_bf16(pf[rbq], vf, av[rbq][dcb]);
      }
    }
  }

  #pragma unroll
  for (int rbq = 0; rbq < 4; ++rbq)
    #pragma unroll
    for (int dcb = 0; dcb < 4; ++dcb)
      #pragma unroll
      for (int e = 0; e < 8; ++e) {
        int Mq = q0 + rbq * 16 + mhi + e;
        int Dc = colbase + w * 64 + dcb * 16 + nloc;
        attended[(size_t)Mq * 512 + Dc] = av[rbq][dcb][e];
      }
}

// ---------------------------------------------------------------------------
// Host launcher
// ---------------------------------------------------------------------------
extern "C" void kernel_launch(void* const* d_in, const int* in_sizes, int n_in,
                              void* d_out, int out_size, void* d_ws, size_t ws_size,
                              hipStream_t stream) {
  (void)in_sizes; (void)n_in; (void)out_size; (void)ws_size;
  const float*     node_features = (const float*)d_in[0];
  const int*       adjacency     = (const int*)d_in[1];
  const long long* edge_index    = (const long long*)d_in[2];
  const float* ne_w = (const float*)d_in[3];
  const float* ne_b = (const float*)d_in[4];
  const float* ne_g = (const float*)d_in[5];
  const float* ne_beta = (const float*)d_in[6];
  const float* ee_w = (const float*)d_in[7];
  const float* ee_b = (const float*)d_in[8];
  const float* ee_g = (const float*)d_in[9];
  const float* ee_beta = (const float*)d_in[10];
  const float* attn_in_w  = (const float*)d_in[11];
  const float* attn_in_b  = (const float*)d_in[12];
  const float* attn_out_w = (const float*)d_in[13];
  const float* attn_out_b = (const float*)d_in[14];
  const float* rc_w1 = (const float*)d_in[15];
  const float* rc_b1 = (const float*)d_in[16];
  const float* rc_w2 = (const float*)d_in[17];
  const float* rc_b2 = (const float*)d_in[18];
  const float* ce_w1 = (const float*)d_in[19];
  const float* ce_b1 = (const float*)d_in[20];
  const float* ce_w2 = (const float*)d_in[21];
  const float* ce_b2 = (const float*)d_in[22];

  float* out = (float*)d_out;
  float* rel_out  = out;                       // [65536, 6]
  float* ctx_out  = out + 393216;              // [4096, 512]
  float* attn_out = out + 393216 + 2097152;    // [1, 4096, 4096]

  // ---- f32 workspace ----
  float* ws = (float*)d_ws;
  float* node_hidden  = ws;                          // 2,097,152
  float* qkv          = node_hidden  + 2097152;      // 6,291,456
  float* edge_hidden  = qkv          + 6291456;      // 33,554,432
  float* rc_hidden    = edge_hidden  + 33554432;     // 16,777,216
  float* attended     = rc_hidden    + 16777216;     // 2,097,152
  float* attended_out = attended     + 2097152;      // 2,097,152
  float* h1           = attended_out + 2097152;      // 1,048,576
  // ---- bf16 workspace (after 63,963,136 floats) ----
  __bf16* bfp    = (__bf16*)(ws + 63963136);
  __bf16* nf_bf  = bfp;                 bfp += 2097152;   // node_features
  __bf16* nh_bf  = bfp;                 bfp += 2097152;   // node_hidden
  __bf16* eh_bf  = bfp;                 bfp += 33554432;  // edge_hidden
  __bf16* at_bf  = bfp;                 bfp += 2097152;   // attended
  __bf16* ao_bf  = bfp;                 bfp += 2097152;   // attended_out
  __bf16* h1_bf  = bfp;                 bfp += 1048576;   // h1
  __bf16* wt_ne  = bfp;                 bfp += 262144;    // [512][512]
  __bf16* wt_ain = bfp;                 bfp += 786432;    // [1536][512]
  __bf16* wt_ee  = bfp;                 bfp += 524288;    // [512][1024]
  __bf16* wt_rc1 = bfp;                 bfp += 131072;    // [256][512]
  __bf16* wt_aow = bfp;                 bfp += 262144;    // [512][512]
  __bf16* wt_ce1 = bfp;                 bfp += 131072;    // [256][512]
  __bf16* wt_ce2 = bfp;                 bfp += 131072;    // [512][256]

  const long long* srcI = edge_index;
  const long long* dstI = edge_index + 65536;

  // ---- weight transpose+convert (once per launch) ----
  transpose_bf16_kernel<<<1024, 256, 0, stream>>>(ne_w,       wt_ne,  512,  512);
  transpose_bf16_kernel<<<3072, 256, 0, stream>>>(attn_in_w,  wt_ain, 512, 1536);
  transpose_bf16_kernel<<<2048, 256, 0, stream>>>(ee_w,       wt_ee, 1024,  512);
  transpose_bf16_kernel<<< 512, 256, 0, stream>>>(rc_w1,      wt_rc1, 512,  256);
  transpose_bf16_kernel<<<1024, 256, 0, stream>>>(attn_out_w, wt_aow, 512,  512);
  transpose_bf16_kernel<<< 512, 256, 0, stream>>>(ce_w1,      wt_ce1, 512,  256);
  transpose_bf16_kernel<<< 512, 256, 0, stream>>>(ce_w2,      wt_ce2, 256,  512);

  // ---- node encoder ----
  f32_to_bf16_kernel<<<2048, 256, 0, stream>>>(node_features, nf_bf, 2097152);
  gemm_bf16_v2<0><<<dim3(4, 32), 256, 0, stream>>>(
      nf_bf, wt_ne, ne_b, node_hidden, 4096, 512, 512, 1, nullptr, nullptr, nullptr);
  ln_kernel<<<512, 256, 0, stream>>>(node_hidden, ne_g, ne_beta, 4096);
  f32_to_bf16_kernel<<<2048, 256, 0, stream>>>(node_hidden, nh_bf, 2097152);

  // ---- qkv projection ----
  gemm_bf16_v2<0><<<dim3(12, 32), 256, 0, stream>>>(
      nh_bf, wt_ain, attn_in_b, qkv, 4096, 512, 1536, 0, nullptr, nullptr, nullptr);

  // ---- edge encoder (gathered concat rows) ----
  gemm_bf16_v2<1><<<dim3(4, 512), 256, 0, stream>>>(
      nullptr, wt_ee, ee_b, edge_hidden, 65536, 1024, 512, 1, srcI, dstI, nh_bf);
  ln_kernel<<<8192, 256, 0, stream>>>(edge_hidden, ee_g, ee_beta, 65536);
  f32_to_bf16_kernel<<<32768, 256, 0, stream>>>(edge_hidden, eh_bf, 33554432);

  // ---- attention ----
  zero_kernel<<<65536, 256, 0, stream>>>(attn_out, (size_t)4096 * 4096);
  attn_kernel<<<dim3(64, 2), 128, 0, stream>>>(qkv, adjacency, attended, attn_out);
  f32_to_bf16_kernel<<<2048, 256, 0, stream>>>(attended, at_bf, 2097152);

  // ---- out projection ----
  gemm_bf16_v2<0><<<dim3(4, 32), 256, 0, stream>>>(
      at_bf, wt_aow, attn_out_b, attended_out, 4096, 512, 512, 0,
      nullptr, nullptr, nullptr);

  // ---- relation classifier ----
  gemm_bf16_v2<0><<<dim3(2, 512), 256, 0, stream>>>(
      eh_bf, wt_rc1, rc_b1, rc_hidden, 65536, 512, 256, 1, nullptr, nullptr, nullptr);
  relcls_kernel<<<8192, 256, 0, stream>>>(rc_hidden, rc_w2, rc_b2, rel_out);

  // ---- context encoder ----
  f32_to_bf16_kernel<<<2048, 256, 0, stream>>>(attended_out, ao_bf, 2097152);
  gemm_bf16_v2<0><<<dim3(2, 32), 256, 0, stream>>>(
      ao_bf, wt_ce1, ce_b1, h1, 4096, 512, 256, 1, nullptr, nullptr, nullptr);
  f32_to_bf16_kernel<<<1024, 256, 0, stream>>>(h1, h1_bf, 1048576);
  gemm_bf16_v2<0><<<dim3(4, 32), 256, 0, stream>>>(
      h1_bf, wt_ce2, ce_b2, ctx_out, 4096, 256, 512, 0, nullptr, nullptr, nullptr);
}
